// MultiHeadAttentionQuantum_65481071409457
// MI455X (gfx1250) — matrix operand order repaired and checked by reading, not compiled, not run on hardware
//
#include <hip/hip_runtime.h>
#include <hip/hip_bf16.h>

typedef __attribute__((ext_vector_type(16))) _Float16 v16h;
typedef __attribute__((ext_vector_type(8)))  _Float16 v8h;
typedef __attribute__((ext_vector_type(8)))  float    v8f;

#define B_  4
#define S_  2048
#define E_  512
#define H_  8
#define DK_ 64

#define WMMA_F16(A, Bv, C) \
  __builtin_amdgcn_wmma_f32_16x16x32_f16(false, (A), false, (Bv), (short)0, (C), false, false)

// ---- CDNA5 async global->LDS copy (ASYNCcnt) --------------------------------
// LDS byte offset is the low 32 bits of the flat shared-aperture address
// 16B per lane per instruction.
static __device__ __forceinline__ void async_ld16(const _Float16* lds_p, const _Float16* g) {
  const unsigned off = (unsigned)(uintptr_t)lds_p;
  asm volatile("global_load_async_to_lds_b128 %0, %1, off" :: "v"(off), "v"(g) : "memory");
}
static __device__ __forceinline__ void wait_async0() {
  asm volatile("s_wait_asynccnt 0" ::: "memory");
}

// ---- DPP 16-lane row reductions (VALU only, no LDS) -------------------------
template <int CTRL>
static __device__ __forceinline__ float dpp_rot(float v) {
  return __int_as_float(__builtin_amdgcn_update_dpp(
      0, __float_as_int(v), CTRL, 0xF, 0xF, true));
}
static __device__ __forceinline__ float row_max16(float v) {
  v = fmaxf(v, dpp_rot<0x121>(v));
  v = fmaxf(v, dpp_rot<0x122>(v));
  v = fmaxf(v, dpp_rot<0x124>(v));
  v = fmaxf(v, dpp_rot<0x128>(v));
  return v;
}
static __device__ __forceinline__ float row_sum16(float v) {
  v += dpp_rot<0x121>(v);
  v += dpp_rot<0x122>(v);
  v += dpp_rot<0x124>(v);
  v += dpp_rot<0x128>(v);
  return v;
}

// A-fragment (16x32 f16): lanes 0-15 hold K {0..15}; lanes 16-31 {16..31}.
static __device__ __forceinline__ v16h load_a_frag(const _Float16* __restrict__ rowp, int lane) {
  const int b0 = (lane & 16) ? 16 : 0;
  v8h lo = *(const v8h*)(rowp + b0);
  v8h hh = *(const v8h*)(rowp + b0 + 8);
  v16h r;
#pragma unroll
  for (int i = 0; i < 8; ++i) { r[i] = lo[i]; r[i + 8] = hh[i]; }
  return r;
}

// B-fragment: 16 contiguous K halfs per lane (two 16B LDS loads).
static __device__ __forceinline__ v16h lds_v16(const _Float16* p) {
  v8h a = *(const v8h*)p;
  v8h b = *(const v8h*)(p + 8);
  v16h r;
#pragma unroll
  for (int i = 0; i < 8; ++i) { r[i] = a[i]; r[i + 8] = b[i]; }
  return r;
}

// Pass 1: qkv = cos(x + theta) in f16, two layouts via LDS-tiled transpose:
//   P  [b,h,s,d] (Q A-frags, K rows), PT [b,h,d,s] (V rows, contiguous keys)
__global__ __launch_bounds__(256) void qkv_cos_kernel(const float* __restrict__ x,
                                                      const float* __restrict__ theta,
                                                      _Float16* __restrict__ P,
                                                      _Float16* __restrict__ PT) {
  __shared__ _Float16 t[64 * 33];
  const int blk = blockIdx.x;      // 0..2047
  const int st  = blk & 63;
  const int h   = (blk >> 6) & 7;
  const int b   = blk >> 9;
  const int sbase = st * 32;
  const size_t bh = (size_t)(b * H_ + h);
#pragma unroll
  for (int k = 0; k < 8; ++k) {
    const int i  = threadIdx.x + k * 256;
    const int d  = i & 63;
    const int sl = i >> 6;
    const int s  = sbase + sl;
    const float v = __cosf(x[((size_t)(b * S_ + s)) * E_ + h * DK_ + d] + theta[d]);
    const _Float16 hv = (_Float16)v;
    P[(bh * S_ + s) * DK_ + d] = hv;
    t[d * 33 + sl] = hv;
  }
  __syncthreads();
#pragma unroll
  for (int k = 0; k < 8; ++k) {
    const int i  = threadIdx.x + k * 256;
    const int sl = i & 31;
    const int d  = i >> 5;
    PT[(bh * DK_ + d) * S_ + sbase + sl] = t[d * 33 + sl];
  }
}

__global__ void wconv_kernel(const float* __restrict__ W, _Float16* __restrict__ Wh) {
  const int idx = blockIdx.x * blockDim.x + threadIdx.x;
  Wh[idx] = (_Float16)W[idx];
}

// Pass 2: fused flash attention. 4 waves/block share one (b,h); K/V tiles are
// double-buffered in LDS via async global->LDS DMA (read once per block, not
// once per wave). 16 WMMAs + DPP online softmax per 64-key iteration.
__global__ __launch_bounds__(128) void flash_attn_kernel(const _Float16* __restrict__ P,
                                                         const _Float16* __restrict__ PT,
                                                         _Float16* __restrict__ AO) {
  // Row stride 72 halfs (36 dwords): 16 fragment rows map to 16 distinct bank
  // groups (gcd(36,64)=4), avoiding conflicts on the 128B-strided reads.
  __shared__ __align__(16) _Float16 ldsK[2][64 * 72];
  __shared__ __align__(16) _Float16 ldsV[2][64 * 72];
  __shared__ __align__(16) _Float16 ldsP[4][16 * 64];

  const int lane = threadIdx.x & 31;
  const int wid  = threadIdx.x >> 5;
  const int wg   = blockIdx.x * 4 + wid;   // 4 consecutive q-tiles, same (b,h)
  const int qt   = wg & 127;
  const int h    = (wg >> 7) & 7;
  const int b    = wg >> 10;
  const int qbase = qt * 16;

  const _Float16* qk = P  + ((size_t)(b * H_ + h)) * S_ * DK_;  // [s][d]
  const _Float16* vt = PT + ((size_t)(b * H_ + h)) * DK_ * S_;  // [d][s]

  const int col  = lane & 15;
  const int hi   = (lane & 16) ? 1 : 0;
  const int hi16 = hi * 16;

  // Issue one 64x64 K tile + 64x64 V tile into LDS buffer `buf` (block-wide:
  // each wave contributes 8 async b128s; 512B moved per instruction).
  auto stage = [&](int buf, int kb) {
    const _Float16* kbp = qk + (size_t)kb * DK_;
#pragma unroll
    for (int j = 0; j < 4; ++j) {
      const int c   = wid * 32 + j * 128 + lane;  // chunk 0..511 (16B each)
      const int row = c >> 3;
      const int ch  = c & 7;
      async_ld16(&ldsK[buf][row * 72 + ch * 8], kbp + (size_t)c * 8);
      async_ld16(&ldsV[buf][row * 72 + ch * 8], vt + (size_t)row * S_ + kb + ch * 8);
    }
  };

  // Q tile A-fragments; fold softmax scale 1/sqrt(64) into Q (exact pow2)
  const _Float16* qrow = qk + (size_t)(qbase + col) * DK_;
  v16h aq_lo = load_a_frag(qrow, lane);
  v16h aq_hi = load_a_frag(qrow + 32, lane);
#pragma unroll
  for (int i = 0; i < 16; ++i) {
    aq_lo[i] = aq_lo[i] * (_Float16)0.125f;
    aq_hi[i] = aq_hi[i] * (_Float16)0.125f;
  }

  stage(0, 0);  // prologue: first tile in flight while we init accumulators

  v8f o[4] = {}, sc[4];
  float m[8], l[8];
#pragma unroll
  for (int r = 0; r < 8; ++r) { m[r] = -1e30f; l[r] = 0.0f; }

  for (int kb = 0; kb < S_; kb += 64) {
    const int buf = (kb >> 6) & 1;
    wait_async0();        // this wave's tile-share landed
    __syncthreads();      // all waves' shares visible
    if (kb + 64 < S_) stage(buf ^ 1, kb + 64);  // uniform branch; overlaps compute

    // ---- scores: S[16x64] = Q(16x64) * K(64x64)^T from LDS ----
#pragma unroll
    for (int j = 0; j < 4; ++j) {
      const _Float16* kr = &ldsK[buf][(16 * j + col) * 72];
      const v16h bl = lds_v16(kr + hi16);
      const v16h bh2 = lds_v16(kr + 32 + hi16);
      v8f s = {};
      s = WMMA_F16(aq_lo, bl, s);
      s = WMMA_F16(aq_hi, bh2, s);
      sc[j] = s;
    }

    // ---- DPP online softmax: rows in VGPR slot r, cols across 16 lanes ----
    float alpha[8];
#pragma unroll
    for (int r = 0; r < 8; ++r) {
      float mx = fmaxf(fmaxf(sc[0][r], sc[1][r]), fmaxf(sc[2][r], sc[3][r]));
      mx = row_max16(mx);
      const float mn = fmaxf(m[r], mx);
      const float al = __expf(m[r] - mn);
      m[r] = mn;
      alpha[r] = al;
      const float p0 = __expf(sc[0][r] - mn);
      const float p1 = __expf(sc[1][r] - mn);
      const float p2 = __expf(sc[2][r] - mn);
      const float p3 = __expf(sc[3][r] - mn);
      sc[0][r] = p0; sc[1][r] = p1; sc[2][r] = p2; sc[3][r] = p3;
      float rs = (p0 + p1) + (p2 + p3);
      rs = row_sum16(rs);
      l[r] = l[r] * al + rs;
    }
#pragma unroll
    for (int j = 0; j < 4; ++j)
#pragma unroll
      for (int r = 0; r < 8; ++r) o[j][r] *= alpha[r];

    // ---- P: D-layout -> A-layout through per-wave LDS tile (16x64) ----
    const int roff = hi * 8;
#pragma unroll
    for (int r = 0; r < 8; ++r) {
      _Float16* lp = &ldsP[wid][(r + roff) * 64 + col];
      lp[0]  = (_Float16)sc[0][r];
      lp[16] = (_Float16)sc[1][r];
      lp[32] = (_Float16)sc[2][r];
      lp[48] = (_Float16)sc[3][r];
    }
    __syncthreads();
    v16h pa_lo, pa_hi;
    {
      const _Float16* rp = &ldsP[wid][(size_t)(lane & 15) * 64];
      const int b0 = hi * 16;
      v8h l0 = *(const v8h*)(rp + b0);
      v8h h0 = *(const v8h*)(rp + b0 + 8);
      v8h l1 = *(const v8h*)(rp + 32 + b0);
      v8h h1 = *(const v8h*)(rp + 32 + b0 + 8);
#pragma unroll
      for (int i = 0; i < 8; ++i) {
        pa_lo[i] = l0[i]; pa_lo[i + 8] = h0[i];
        pa_hi[i] = l1[i]; pa_hi[i + 8] = h1[i];
      }
    }
    __syncthreads();

    // ---- O += P(16x64) * V(64x64) from LDS (V^T rows: contiguous keys) ----
#pragma unroll
    for (int j = 0; j < 4; ++j) {
      const _Float16* vr = &ldsV[buf][(16 * j + col) * 72];
      const v16h bv = lds_v16(vr + hi16);
      const v16h bw = lds_v16(vr + 32 + hi16);
      o[j] = WMMA_F16(pa_lo, bv, o[j]);
      o[j] = WMMA_F16(pa_hi, bw, o[j]);
    }
  }

  // ---- normalize and store attention output as f16 in [b*S+s][E] layout ----
  float inv[8];
#pragma unroll
  for (int r = 0; r < 8; ++r) inv[r] = 1.0f / l[r];
  _Float16* outp = AO + ((size_t)b * S_ + qbase) * E_ + h * DK_;
#pragma unroll
  for (int r = 0; r < 8; ++r) {
    _Float16* rowp = outp + (size_t)(r + hi * 8) * E_;
    rowp[0 * 16 + col] = (_Float16)(o[0][r] * inv[r]);
    rowp[1 * 16 + col] = (_Float16)(o[1][r] * inv[r]);
    rowp[2 * 16 + col] = (_Float16)(o[2][r] * inv[r]);
    rowp[3 * 16 + col] = (_Float16)(o[3][r] * inv[r]);
  }
}

// Pass 3: out[8192,512] = AO[8192,512] * W^T + b   (W rows contiguous over e)
__global__ __launch_bounds__(128) void proj_gemm_kernel(const _Float16* __restrict__ A,
                                                        const _Float16* __restrict__ Wh,
                                                        const float* __restrict__ bias,
                                                        float* __restrict__ out) {
  const int lane = threadIdx.x & 31;
  const int wid  = threadIdx.x >> 5;
  const int wg   = blockIdx.x * 4 + wid;
  const int rt   = wg & 511;
  const int ct   = wg >> 9;
  const int col  = lane & 15;
  const int hi16 = (lane & 16) ? 16 : 0;
  const int rowbase = rt * 16;
  const int colbase = ct * 64;

  const _Float16* arow = A + (size_t)(rowbase + col) * E_;
  v8f c0 = {}, c1 = {}, c2 = {}, c3 = {};
  for (int e0 = 0; e0 < E_; e0 += 32) {
    const v16h a  = load_a_frag(arow + e0, lane);
    const v16h b0 = *(const v16h*)(Wh + (size_t)(colbase + 0 * 16 + col) * E_ + e0 + hi16);
    const v16h b1 = *(const v16h*)(Wh + (size_t)(colbase + 1 * 16 + col) * E_ + e0 + hi16);
    const v16h b2 = *(const v16h*)(Wh + (size_t)(colbase + 2 * 16 + col) * E_ + e0 + hi16);
    const v16h b3 = *(const v16h*)(Wh + (size_t)(colbase + 3 * 16 + col) * E_ + e0 + hi16);
    if (e0 + 32 < E_) {
      __builtin_prefetch(arow + e0 + 32, 0, 3);
    }
    c0 = WMMA_F16(a, b0, c0);
    c1 = WMMA_F16(a, b1, c1);
    c2 = WMMA_F16(a, b2, c2);
    c3 = WMMA_F16(a, b3, c3);
  }
  const int roff = (lane & 16) ? 8 : 0;
#pragma unroll
  for (int r = 0; r < 8; ++r) {
    float* orow = out + (size_t)(rowbase + r + roff) * E_ + colbase;
    orow[0 * 16 + col] = c0[r] + bias[colbase + 0 * 16 + col];
    orow[1 * 16 + col] = c1[r] + bias[colbase + 1 * 16 + col];
    orow[2 * 16 + col] = c2[r] + bias[colbase + 2 * 16 + col];
    orow[3 * 16 + col] = c3[r] + bias[colbase + 3 * 16 + col];
  }
}

extern "C" void kernel_launch(void* const* d_in, const int* in_sizes, int n_in,
                              void* d_out, int out_size, void* d_ws, size_t ws_size,
                              hipStream_t stream) {
  const float* x     = (const float*)d_in[0];  // [B,S,E]
  const float* theta = (const float*)d_in[1];  // [DK]
  const float* W     = (const float*)d_in[2];  // [E,E]
  const float* bias  = (const float*)d_in[3];  // [E]
  float* out = (float*)d_out;                  // [B,S,E] f32

  char* ws = (char*)d_ws;
  // Workspace: P 8MiB | PT 8MiB | AO 8MiB | Wh 0.5MiB  (all fit in L2)
  _Float16* P  = (_Float16*)(ws);
  _Float16* PT = (_Float16*)(ws + (size_t)8 * 1024 * 1024);
  _Float16* AO = (_Float16*)(ws + (size_t)16 * 1024 * 1024);
  _Float16* Wh = (_Float16*)(ws + (size_t)24 * 1024 * 1024);

  qkv_cos_kernel<<<2048, 256, 0, stream>>>(x, theta, P, PT);
  wconv_kernel<<<1024, 256, 0, stream>>>(W, Wh);
  flash_attn_kernel<<<1024, 128, 0, stream>>>(P, PT, AO);
  proj_gemm_kernel<<<1024, 128, 0, stream>>>(AO, Wh, bias, out);
}
